// RelativePositionSDPA_48670569398462
// MI455X (gfx1250) — compile-verified, matrix operand order checked by reading, not run on hardware
//
#include <hip/hip_runtime.h>

// ---------------- problem constants ----------------
#define S_LEN 1024   // S == S_kv
#define KD    64     // head dim
#define HN    16     // num heads
#define NHB   64     // batch*heads
#define RMAX  2048   // R's max_seq_len
static constexpr float SCALE = 0.125f;  // 64^-0.5

typedef __attribute__((ext_vector_type(16))) _Float16 v16h;
typedef __attribute__((ext_vector_type(8)))  _Float16 v8h;
typedef __attribute__((ext_vector_type(8)))  float    v8f;

// Load a 16x32 f16 A/B fragment. rowp points at element k=0 of THIS lane's row
// (row = lane&15 chosen by caller). Slot->k mapping: k in {h*8..h*8+7} U
// {16+h*8..16+h*8+7}, h = lane>>4. Same mapping used for every A and B operand,
// so the contraction index pairing is self-consistent.
__device__ __forceinline__ v16h load_frag16(const _Float16* rowp, int hh) {
  v8h a = *reinterpret_cast<const v8h*>(rowp + hh * 8);
  v8h b = *reinterpret_cast<const v8h*>(rowp + 16 + hh * 8);
  v16h r;
#pragma unroll
  for (int i = 0; i < 8; ++i) { r[i] = a[i]; r[i + 8] = b[i]; }
  return r;
}

__device__ __forceinline__ v8f wmma_f16(v16h a, v16h b, v8f c) {
  return __builtin_amdgcn_wmma_f32_16x16x32_f16(false, a, false, b,
                                                (short)0, c, false, false);
}

// ---------------- prep kernels (fp32 -> f16 staging) ----------------
__global__ void prep_q(const float* __restrict__ q, const float* __restrict__ ub,
                       const float* __restrict__ vb, _Float16* __restrict__ qu,
                       _Float16* __restrict__ qv) {
  size_t i = (size_t)blockIdx.x * 256 + threadIdx.x;
  if (i >= (size_t)NHB * S_LEN * KD) return;
  int k = (int)(i % KD);
  int b = (int)(i / ((size_t)S_LEN * KD));
  int hd = b % HN;                       // q.reshape(N,H,S,K): head = b % H
  float qs = q[i] * SCALE;
  qu[i] = (_Float16)(qs + ub[hd * KD + k]);
  qv[i] = (_Float16)(qs + vb[hd * KD + k]);
}

__global__ void prep_cvt(const float* __restrict__ in, _Float16* __restrict__ out,
                         int n) {
  size_t i = (size_t)blockIdx.x * 256 + threadIdx.x;
  if (i < (size_t)n) out[i] = (_Float16)in[i];
}

// vT[b][k][t] = values[b][t][k]   (so P@V B-fragments load contiguously)
__global__ void prep_vT(const float* __restrict__ vals, _Float16* __restrict__ vT) {
  size_t i = (size_t)blockIdx.x * 256 + threadIdx.x;
  if (i >= (size_t)NHB * KD * S_LEN) return;
  int t = (int)(i % S_LEN);
  size_t rest = i / S_LEN;
  int k = (int)(rest % KD);
  int b = (int)(rest / KD);
  vT[i] = (_Float16)(vals[((size_t)b * S_LEN + t) * KD + k]);
}

// r16[h][j][k] = R[h][j][k] for j < S_LEN (R has RMAX rows per head)
__global__ void prep_r(const float* __restrict__ R, _Float16* __restrict__ r16) {
  size_t i = (size_t)blockIdx.x * 256 + threadIdx.x;
  if (i >= (size_t)HN * S_LEN * KD) return;
  int hd = (int)(i / ((size_t)S_LEN * KD));
  size_t rem = i % ((size_t)S_LEN * KD);
  r16[i] = (_Float16)(R[(size_t)hd * RMAX * KD + rem]);
}

// ---------------- fused rel-pos flash attention ----------------
// One workgroup = 4 wave32 = 128 threads, handles (b, 16 query rows).
__global__ __launch_bounds__(128) void xl_attn(
    const _Float16* __restrict__ qu16, const _Float16* __restrict__ qv16,
    const _Float16* __restrict__ k16,  const _Float16* __restrict__ v16T,
    const _Float16* __restrict__ r16,  float* __restrict__ out) {
  __shared__ alignas(16) _Float16 Pl[17][1040];       // P rows s0..s0+16 (f16)
  __shared__ alignas(16) _Float16 pScr[4][16][32];    // per-wave p re-layout
  __shared__ alignas(16) float    Ol[4][16][64];      // per-wave partial O
  __shared__ float ml[4][16], ll[4][16];
  __shared__ float qx[KD];

  const int b  = blockIdx.y;
  const int hd = b % HN;
  const int s0 = blockIdx.x * 16;
  const int tid = threadIdx.x;
  const int w  = tid >> 5;
  const int l  = tid & 31;
  const int hh = l >> 4;        // lane half (k-split of fragments)
  const int nn = l & 15;        // row (A) / col (B,C,D) index

  // ---------- Phase 1: P[s0..s0+15][0..1023] = qv @ R^T via WMMA ----------
  {
    const _Float16* qvrow = qv16 + ((size_t)b * S_LEN + s0 + nn) * KD;
    v16h qvA0 = load_frag16(qvrow, hh);
    v16h qvA1 = load_frag16(qvrow + 32, hh);
    for (int jt = w; jt < 64; jt += 4) {
      int j0 = jt * 16;
      const _Float16* rrow = r16 + ((size_t)hd * S_LEN + j0 + nn) * KD;
      if (jt + 4 < 64)  // prefetch this lane's R row of the next tile (WGP scope)
        __builtin_prefetch(rrow + 4 * 16 * KD, 0, 3);
      v16h rb0 = load_frag16(rrow, hh);
      v16h rb1 = load_frag16(rrow + 32, hh);
      v8f d = {};
      d = wmma_f16(qvA0, rb0, d);
      d = wmma_f16(qvA1, rb1, d);
#pragma unroll
      for (int r = 0; r < 8; ++r)
        Pl[r + hh * 8][j0 + nn] = (_Float16)d[r];
    }
  }
  // ---------- extra P row s0+16 (needed by shift's upper branch) ----------
  {
    bool valid = (s0 + 16) < S_LEN;
    int  ser = valid ? (s0 + 16) : (S_LEN - 1);
    if (tid < KD) qx[tid] = (float)qv16[((size_t)b * S_LEN + ser) * KD + tid];
    __syncthreads();
    for (int j = tid; j < S_LEN; j += 128) {
      const v8h* rr =
          reinterpret_cast<const v8h*>(r16 + ((size_t)hd * S_LEN + j) * KD);
      float acc = 0.f;
#pragma unroll
      for (int kk = 0; kk < 8; ++kk) {   // 8 x b128 vector loads
        v8h rv = rr[kk];
#pragma unroll
        for (int e = 0; e < 8; ++e) acc += qx[kk * 8 + e] * (float)rv[e];
      }
      Pl[16][j] = valid ? (_Float16)acc : (_Float16)0.f;
    }
  }
  __syncthreads();

  // ---------- Phase 2: flash attention, each wave owns 32-col chunks ------
  const _Float16* qurow = qu16 + ((size_t)b * S_LEN + s0 + nn) * KD;
  v16h quA0 = load_frag16(qurow, hh);
  v16h quA1 = load_frag16(qurow + 32, hh);

  float m_r[8], l_r[8];
  v8f acc[4];
#pragma unroll
  for (int r = 0; r < 8; ++r) { m_r[r] = -3.0e38f; l_r[r] = 0.f; }
#pragma unroll
  for (int vt = 0; vt < 4; ++vt) acc[vt] = (v8f){};

  for (int c = 0; c < 8; ++c) {
    const int tc = w * 32 + c * 128;        // this wave's 32 key columns

    // prefetch next chunk's K and V rows into near caches (WGP scope)
    if (c + 1 < 8) {
      const int tn = tc + 128;
      __builtin_prefetch(k16 + ((size_t)b * S_LEN + tn + nn) * KD, 0, 3);
      __builtin_prefetch(k16 + ((size_t)b * S_LEN + tn + 16 + nn) * KD, 0, 3);
#pragma unroll
      for (int vt = 0; vt < 4; ++vt)
        __builtin_prefetch(
            v16T + ((size_t)b * KD + vt * 16 + nn) * S_LEN + tn, 0, 3);
    }

    float sc[2][8];
#pragma unroll
    for (int tt = 0; tt < 2; ++tt) {
      int t0 = tc + tt * 16;
      const _Float16* krow = k16 + ((size_t)b * S_LEN + t0 + nn) * KD;
      v16h kb0 = load_frag16(krow, hh);
      v16h kb1 = load_frag16(krow + 32, hh);
      v8f d = {};
      d = wmma_f16(quA0, kb0, d);
      d = wmma_f16(quA1, kb1, d);
#pragma unroll
      for (int r = 0; r < 8; ++r) {
        int srel = r + hh * 8;
        int s = s0 + srel;
        int t = t0 + nn;
        float pos;
        if (t <= s)            pos = (float)Pl[srel][(S_LEN - 1) - s + t];
        else if (t == s + 1)   pos = 0.f;
        else                   pos = (float)Pl[srel + 1][t - s - 2];
        sc[tt][r] = d[r] + pos;
      }
    }
    // online softmax over this 32-col chunk (rows live in 16-lane groups)
    float alpha[8];
#pragma unroll
    for (int r = 0; r < 8; ++r) {
      float rmax = fmaxf(sc[0][r], sc[1][r]);
      rmax = fmaxf(rmax, __shfl_xor(rmax, 1, 32));
      rmax = fmaxf(rmax, __shfl_xor(rmax, 2, 32));
      rmax = fmaxf(rmax, __shfl_xor(rmax, 4, 32));
      rmax = fmaxf(rmax, __shfl_xor(rmax, 8, 32));
      float mnew = fmaxf(m_r[r], rmax);
      alpha[r] = __expf(m_r[r] - mnew);
      m_r[r] = mnew;
      sc[0][r] = __expf(sc[0][r] - mnew);
      sc[1][r] = __expf(sc[1][r] - mnew);
      float ps = sc[0][r] + sc[1][r];
      ps += __shfl_xor(ps, 1, 32);
      ps += __shfl_xor(ps, 2, 32);
      ps += __shfl_xor(ps, 4, 32);
      ps += __shfl_xor(ps, 8, 32);
      l_r[r] = l_r[r] * alpha[r] + ps;
    }
#pragma unroll
    for (int vt = 0; vt < 4; ++vt)
#pragma unroll
      for (int r = 0; r < 8; ++r) acc[vt][r] *= alpha[r];

    // re-layout p (C/D layout) -> A fragment via wave-private LDS
#pragma unroll
    for (int tt = 0; tt < 2; ++tt)
#pragma unroll
      for (int r = 0; r < 8; ++r)
        pScr[w][r + hh * 8][tt * 16 + nn] = (_Float16)sc[tt][r];
    asm volatile("" ::: "memory");          // keep store->reload ordered
    v16h pA = load_frag16(&pScr[w][nn][0], hh);

#pragma unroll
    for (int vt = 0; vt < 4; ++vt) {
      const _Float16* vrow = v16T + ((size_t)b * KD + vt * 16 + nn) * S_LEN + tc;
      v16h vb = load_frag16(vrow, hh);
      acc[vt] = wmma_f16(pA, vb, acc[vt]);
    }
  }

  // ---------- epilogue: merge the 4 waves' partial softmaxes -------------
#pragma unroll
  for (int vt = 0; vt < 4; ++vt)
#pragma unroll
    for (int r = 0; r < 8; ++r)
      Ol[w][r + hh * 8][vt * 16 + nn] = acc[vt][r];
  if (nn == 0) {
#pragma unroll
    for (int r = 0; r < 8; ++r) {
      ml[w][r + hh * 8] = m_r[r];
      ll[w][r + hh * 8] = l_r[r];
    }
  }
  __syncthreads();

  for (int idx = tid; idx < 16 * 64; idx += 128) {
    int row = idx >> 6, col = idx & 63;
    float mf = ml[0][row];
    mf = fmaxf(mf, ml[1][row]);
    mf = fmaxf(mf, ml[2][row]);
    mf = fmaxf(mf, ml[3][row]);
    float lf = 0.f, of = 0.f;
#pragma unroll
    for (int w4 = 0; w4 < 4; ++w4) {
      float e = __expf(ml[w4][row] - mf);
      lf += ll[w4][row] * e;
      of += Ol[w4][row][col] * e;
    }
    out[((size_t)b * S_LEN + s0 + row) * KD + col] = of / lf;
  }
}

// ---------------- host launch ----------------
extern "C" void kernel_launch(void* const* d_in, const int* in_sizes, int n_in,
                              void* d_out, int out_size, void* d_ws, size_t ws_size,
                              hipStream_t stream) {
  (void)in_sizes; (void)n_in; (void)out_size; (void)ws_size;
  const float* q  = (const float*)d_in[0];
  const float* k  = (const float*)d_in[1];
  const float* v  = (const float*)d_in[2];
  const float* ub = (const float*)d_in[3];
  const float* vb = (const float*)d_in[4];
  const float* R  = (const float*)d_in[5];
  float* out = (float*)d_out;

  char* ws = (char*)d_ws;
  const size_t Q2 = (size_t)NHB * S_LEN * KD * 2;  // one f16 Q-sized buffer
  _Float16* qu16 = (_Float16*)(ws);
  _Float16* qv16 = (_Float16*)(ws + Q2);
  _Float16* k16  = (_Float16*)(ws + 2 * Q2);
  _Float16* vT16 = (_Float16*)(ws + 3 * Q2);
  _Float16* r16  = (_Float16*)(ws + 4 * Q2);

  const int n  = NHB * S_LEN * KD;
  const int nr = HN * S_LEN * KD;
  prep_q  <<<(n  + 255) / 256, 256, 0, stream>>>(q, ub, vb, qu16, qv16);
  prep_cvt<<<(n  + 255) / 256, 256, 0, stream>>>(k, k16, n);
  prep_vT <<<(n  + 255) / 256, 256, 0, stream>>>(v, vT16);
  prep_r  <<<(nr + 255) / 256, 256, 0, stream>>>(R, r16);

  dim3 grid(S_LEN / 16, NHB);
  xl_attn<<<grid, 128, 0, stream>>>(qu16, qv16, k16, vT16, r16, out);
}
